// GraphSAGEConv_21887153340602
// MI455X (gfx1250) — matrix operand order repaired and checked by reading, not compiled, not run on hardware
//
#include <hip/hip_runtime.h>

// ---- problem constants (match reference) ----
#define N_NODES 50000
#define N_EDGES 800000
#define IN_CH   512
#define OUT_CH  256
#define M_TILES (N_NODES / 16)   // 3125
#define K_STEPS (IN_CH / 32)     // 16

// TDM builtin arity differs between toolchains; the gfx1250 TDM header ships
// only with the amdgpu-toolchain (6-arg builtin), so probe for it.
#if defined(__has_include)
#if __has_include(<hip/amd_detail/amd_gfx1250_TDM.h>)
#define GCN_TDM_6ARG 1
#endif
#endif
#ifndef GCN_TDM_6ARG
#define GCN_TDM_6ARG 0
#endif

typedef __attribute__((ext_vector_type(16))) __bf16 v16bf;
typedef __attribute__((ext_vector_type(8)))  float  v8f;
typedef __attribute__((ext_vector_type(4)))  float  v4f;
typedef __attribute__((ext_vector_type(4)))  unsigned int u32x4;
typedef __attribute__((ext_vector_type(8)))  int    i32x8;
typedef __attribute__((ext_vector_type(4)))  int    i32x4;

// ---------------- kernel 1a: zero out + deg ----------------
__global__ void k_zero(float* __restrict__ out, float* __restrict__ deg) {
    const size_t n_out = (size_t)N_NODES * OUT_CH;
    const size_t total = n_out + N_NODES;
    for (size_t i = (size_t)blockIdx.x * blockDim.x + threadIdx.x; i < total;
         i += (size_t)gridDim.x * blockDim.x) {
        if (i < n_out) out[i] = 0.0f;
        else           deg[i - n_out] = 0.0f;
    }
}

// ---------------- kernel 1b: W [512][256] f32 -> wb [256][512] bf16 ----------------
__global__ void k_wtrans(const float* __restrict__ w, __bf16* __restrict__ wb) {
    int i = blockIdx.x * blockDim.x + threadIdx.x;   // 131072 threads
    if (i >= IN_CH * OUT_CH) return;
    int k = i & (IN_CH - 1);
    int n = i >> 9;                                   // i / IN_CH
    wb[(size_t)n * IN_CH + k] = (__bf16)w[(size_t)k * OUT_CH + n];
}

__device__ inline void cvt4(v16bf& a, int base, v4f f) {
    a[base + 0] = (__bf16)f.x;
    a[base + 1] = (__bf16)f.y;
    a[base + 2] = (__bf16)f.z;
    a[base + 3] = (__bf16)f.w;
}

// Issue one TDM descriptor: copy wb[0..255][k0..k0+31] (2-D tile, row stride 512
// elems, 2-byte elems) into LDS at byte offset lds_addr, packed [256][32].
__device__ inline void tdm_issue_b(const __bf16* wb, int k0, unsigned lds_addr) {
    unsigned long long ga =
        (unsigned long long)(uintptr_t)wb + (unsigned long long)k0 * 2u;

    u32x4 g0 = {};
    g0[0] = 1u;                                   // count=1, user descriptor
    g0[1] = lds_addr;                             // lds_addr[31:0]
    g0[2] = (unsigned)(ga & 0xFFFFFFFFu);         // global_addr[31:0]
    g0[3] = (unsigned)((ga >> 32) & 0x1FFFFFFu)   // global_addr[56:32]
          | (2u << 30);                           // type=2 ("image")

    i32x8 g1 = {};
    g1[0] = (int)(1u << 16);                      // workgroup_mask=0, data_size=1 (2B)
    g1[1] = (int)(((unsigned)IN_CH & 0xFFFFu) << 16);           // tensor_dim0 lo16
    g1[2] = (int)((((unsigned)IN_CH >> 16) & 0xFFFFu)           // tensor_dim0 hi16
          | (((unsigned)OUT_CH & 0xFFFFu) << 16));              // tensor_dim1 lo16
    g1[3] = (int)((((unsigned)OUT_CH >> 16) & 0xFFFFu)          // tensor_dim1 hi16
          | (32u << 16));                                       // tile_dim0 = 32
    g1[4] = (int)(unsigned)OUT_CH;                // tile_dim1 = 256, tile_dim2 = 0
    g1[5] = (int)(unsigned)IN_CH;                 // tensor_dim0_stride = 512 (lo32)
    g1[6] = 0;                                    // stride hi16 / dim1_stride lo16
    g1[7] = 0;

    i32x4 gz = {};
#if GCN_TDM_6ARG
    i32x8 gz8 = {};
    __builtin_amdgcn_tensor_load_to_lds(g0, g1, gz, gz, gz8, 0);
#else
    __builtin_amdgcn_tensor_load_to_lds(g0, g1, gz, gz, 0);
#endif
}

// ---------------- kernel 2: support = X @ W via v_wmma_f32_16x16x32_bf16 ----------------
// Block = 8 waves = 4 M-tiles x 2 N-groups. B tile (32K x 256N bf16 = 16 KB) is
// shared by all waves: TDM double-buffers it into LDS while waves compute.
__global__ __launch_bounds__(256) void k_gemm(const float* __restrict__ x,
                                              const __bf16* __restrict__ wb,
                                              float* __restrict__ support) {
    __shared__ __bf16 sb[2][OUT_CH][32];          // 2 x 16 KB staging buffers

    const int lane   = threadIdx.x & 31;
    const int wave   = threadIdx.x >> 5;
    const int mtile  = blockIdx.x * 4 + (wave >> 1);
    const int ngroup = wave & 1;
    const bool active = (mtile < M_TILES);        // wave-uniform

    const int idx  = lane & 15;
    const int half = lane >> 4;                   // 0: lanes 0-15, 1: lanes 16-31
    const int row  = mtile * 16 + idx;

    const unsigned lds0 = (unsigned)(uintptr_t)&sb[0][0][0];
    const unsigned lds1 = (unsigned)(uintptr_t)&sb[1][0][0];

    v8f acc[8] = {};
    const v4f* xr4 = active ? (const v4f*)(x + (size_t)row * IN_CH) : (const v4f*)x;

    // prologue: prefetch K-step 0 into buffer 0
    if (wave == 0) tdm_issue_b(wb, 0, lds0);

    for (int kk = 0; kk < K_STEPS; ++kk) {
        __builtin_amdgcn_s_wait_tensorcnt(0);     // wave 0: buf[kk&1] landed
        __syncthreads();                          // visible to all waves
        if (wave == 0 && kk + 1 < K_STEPS)        // prefetch next into other buffer
            tdm_issue_b(wb, (kk + 1) * 32, (kk & 1) ? lds0 : lds1);

        if (active) {
            const int k0 = kk * 32;
            // A fragment (16-bit A 16x32 layout): lane holds
            //   K = k0 + 16*half + 0..15 (contiguous, matches B)
            const int c0 = (k0 + 16 * half) >> 2;
            v4f f0 = xr4[c0 + 0];
            v4f f1 = xr4[c0 + 1];
            const int c1 = c0 + 2;
            v4f f2 = xr4[c1 + 0];
            v4f f3 = xr4[c1 + 1];
            v16bf a;
            cvt4(a, 0, f0); cvt4(a, 4, f1); cvt4(a, 8, f2); cvt4(a, 12, f3);

            #pragma unroll
            for (int t = 0; t < 8; ++t) {
                // B fragment (32x16): lanes 0-15 hold K=k0+0..15 (N=col),
                // lanes 16-31 hold K=k0+16..31 -> 16 contiguous bf16 in LDS
                const int col = (ngroup * 8 + t) * 16 + idx;
                v16bf b = *(const v16bf*)&sb[kk & 1][col][16 * half];
                acc[t] = __builtin_amdgcn_wmma_f32_16x16x32_bf16(
                    false, a, false, b, (short)0, acc[t], false, false);
            }
        }
        __syncthreads();                          // readers done before next DMA lands
    }

    if (active) {
        // C/D layout: VGPR r -> M = r (lanes 0-15) or M = 8+r (lanes 16-31), N = idx
        #pragma unroll
        for (int t = 0; t < 8; ++t) {
            const int col = (ngroup * 8 + t) * 16 + idx;
            #pragma unroll
            for (int r = 0; r < 8; ++r) {
                support[(size_t)(mtile * 16 + 8 * half + r) * OUT_CH + col] = acc[t][r];
            }
        }
    }
}

// ---------------- kernel 3: edge scatter-add (one wave per edge) ----------------
__device__ inline void atomicAddF32(float* p, float v) {
    __hip_atomic_fetch_add(p, v, __ATOMIC_RELAXED, __HIP_MEMORY_SCOPE_AGENT);
}

__global__ __launch_bounds__(256) void k_scatter(const float* __restrict__ support,
                                                 const int* __restrict__ rows,
                                                 const int* __restrict__ cols,
                                                 float* __restrict__ out,
                                                 float* __restrict__ deg) {
    const int e = blockIdx.x * 8 + (threadIdx.x >> 5);
    if (e >= N_EDGES) return;
    const int lane = threadIdx.x & 31;
    const int r = rows[e];
    const int c = cols[e];

    const v4f* s4 = (const v4f*)(support + (size_t)c * OUT_CH);
    v4f v0 = s4[lane * 2 + 0];
    v4f v1 = s4[lane * 2 + 1];

    float* o = out + (size_t)r * OUT_CH + lane * 8;
    atomicAddF32(o + 0, v0.x); atomicAddF32(o + 1, v0.y);
    atomicAddF32(o + 2, v0.z); atomicAddF32(o + 3, v0.w);
    atomicAddF32(o + 4, v1.x); atomicAddF32(o + 5, v1.y);
    atomicAddF32(o + 6, v1.z); atomicAddF32(o + 7, v1.w);

    if (lane == 0) atomicAddF32(deg + r, 1.0f);
}

// ---------------- kernel 4: degree normalization ----------------
__global__ __launch_bounds__(256) void k_norm(float* __restrict__ out,
                                              const float* __restrict__ deg) {
    const size_t i = (size_t)blockIdx.x * 256 + threadIdx.x;  // exactly N_NODES*OUT_CH
    const float d = deg[i >> 8];
    out[i] = out[i] / fmaxf(d, 1.0f);
}

extern "C" void kernel_launch(void* const* d_in, const int* in_sizes, int n_in,
                              void* d_out, int out_size, void* d_ws, size_t ws_size,
                              hipStream_t stream) {
    const float* x    = (const float*)d_in[0];
    const float* w    = (const float*)d_in[1];
    const int*   rows = (const int*)d_in[2];
    const int*   cols = (const int*)d_in[3];
    float*       out  = (float*)d_out;

    // workspace layout
    char* ws = (char*)d_ws;
    __bf16* wb      = (__bf16*)(ws);                 // 512*256*2      = 256 KB
    float*  deg     = (float*)(ws + (256u << 10));   // 50000*4        = 200 KB
    float*  support = (float*)(ws + (512u << 10));   // 50000*256*4    = 51.2 MB

    // 1) init + weight transpose/convert
    k_zero<<<4096, 256, 0, stream>>>(out, deg);
    k_wtrans<<<(IN_CH * OUT_CH + 255) / 256, 256, 0, stream>>>(w, wb);

    // 2) GEMM: 4 M-tiles x 2 N-groups per block (8 waves), TDM-staged B
    k_gemm<<<(M_TILES + 3) / 4, 256, 0, stream>>>(x, wb, support);

    // 3) edge aggregation: 8 edges/block
    k_scatter<<<N_EDGES / 8, 256, 0, stream>>>(support, rows, cols, out, deg);

    // 4) normalize
    k_norm<<<N_NODES, 256, 0, stream>>>(out, deg);
}
